// TransformerBlock_49323404427742
// MI455X (gfx1250) — compile-verified, matrix operand order checked by reading, not run on hardware
//
#include <hip/hip_runtime.h>
#include <hip/hip_bf16.h>

// -------- CDNA5 (gfx1250) transformer block: bf16 WMMA + TDM tile staging ---
#define DMODEL 1024
#define NHEADS 16
#define DKH    64
#define DFF    4096
#define BATCH  2
#define SEQ    2048
#define MROWS  (BATCH * SEQ)   // 4096

typedef __bf16 bf16_t;
typedef __attribute__((ext_vector_type(16))) __bf16 v16bf;
typedef __attribute__((ext_vector_type(8)))  float  v8f;
typedef __attribute__((ext_vector_type(4)))  unsigned u32x4;
typedef __attribute__((ext_vector_type(8)))  int      i32x8;
typedef __attribute__((ext_vector_type(4)))  int      i32x4;

#if defined(__gfx1250__) && __has_builtin(__builtin_amdgcn_tensor_load_to_lds)
#define HAVE_TDM 1
#else
#define HAVE_TDM 0
#endif

// bf16 wave32 WMMA fragment (A 16x32, or B as 32x16 transposed).
// Lane l (half=l>>4): slots 0..7 -> K = e + 8*half, slots 8..15 -> K = e+8+8*half
// => two contiguous 16-byte runs per lane.
union Frag {
    v16bf v;
    uint4 q[2];
    bf16_t e[16];
};

__device__ __forceinline__ v8f wmma_bf16(const Frag& a, const Frag& b, v8f c) {
    return __builtin_amdgcn_wmma_f32_16x16x32_bf16(false, a.v, false, b.v,
                                                   (short)0, c, false, false);
}

// ---- Tensor Data Mover: 2D bf16 tile (tile_rows x tile_elems) global->LDS ---
// pad_int_code: pad every (2<<code) dwords; pad_amt_code: by (code+1) dwords.
__device__ __forceinline__ void tdm_load_2d(const bf16_t* gsrc, void* ldst,
                                            unsigned tile_elems, unsigned tile_rows,
                                            unsigned long long row_stride_elems,
                                            unsigned pad_int_code, unsigned pad_amt_code)
{
#if HAVE_TDM
    unsigned long long ga = (unsigned long long)gsrc;
    unsigned lds = (unsigned)(unsigned long long)ldst;
    u32x4 g0 = { 1u,                                   // count=1, user mode
                 lds,                                  // lds_addr
                 (unsigned)ga,                         // global_addr lo
                 (unsigned)((ga >> 32) & 0x1FFFFFFull) | 0x80000000u }; // hi | type=2
    unsigned d0 = (1u << 16)                 // data_size = 2 bytes
                | (1u << 20)                 // pad_enable
                | (pad_int_code << 22) | (pad_amt_code << 25);
    unsigned d1 = (tile_elems & 0xFFFFu) << 16;                          // tensor_dim0 lo
    unsigned d2 = ((tile_elems >> 16) & 0xFFFFu) | ((tile_rows & 0xFFFFu) << 16); // dim0 hi|dim1 lo
    unsigned d3 = ((tile_rows >> 16) & 0xFFFFu) | ((tile_elems & 0xFFFFu) << 16); // dim1 hi|tile_dim0
    unsigned d4 = (tile_rows & 0xFFFFu);                                 // tile_dim1 (tile_dim2=0)
    unsigned d5 = (unsigned)(row_stride_elems & 0xFFFFFFFFull);          // dim0_stride lo
    unsigned d6 = (unsigned)((row_stride_elems >> 32) & 0xFFFFull);      // dim0_stride hi
    i32x8 g1 = { (int)d0, (int)d1, (int)d2, (int)d3, (int)d4, (int)d5, (int)d6, 0 };
    i32x4 z4 = { 0, 0, 0, 0 };
#if __clang_major__ >= 23
    i32x8 z8 = { 0, 0, 0, 0, 0, 0, 0, 0 };
    __builtin_amdgcn_tensor_load_to_lds(g0, g1, z4, z4, z8, 0);
#else
    __builtin_amdgcn_tensor_load_to_lds(g0, g1, z4, z4, 0);
#endif
#else
    (void)gsrc; (void)ldst; (void)tile_elems; (void)tile_rows;
    (void)row_stride_elems; (void)pad_int_code; (void)pad_amt_code;
#endif
}

// Manual staging fallback: rows x (2*rowdw) bf16, LDS stride in elements.
__device__ __forceinline__ void stage_tile(bf16_t* dst, int lds_stride, int rowdw,
                                           const bf16_t* src, size_t src_stride,
                                           int rows, int t)
{
    const int total = rows * rowdw;
    for (int u = t; u < total; u += 256) {
        int r = u / rowdw, c = u % rowdw;
        *(unsigned*)(dst + r * lds_stride + 2 * c) =
            *(const unsigned*)(src + (size_t)r * src_stride + 2 * c);
    }
}

// ---------------------------------------------------------------- LayerNorm --
__global__ __launch_bounds__(256)
void ln_bf16_kernel(const float* __restrict__ x, const float* __restrict__ g,
                    const float* __restrict__ be, bf16_t* __restrict__ out)
{
    __shared__ float red[16];
    const int row = blockIdx.x;
    const int t   = threadIdx.x;
    const float* xr = x + (size_t)row * DMODEL;
    float4 v = ((const float4*)xr)[t];

    float s = v.x + v.y + v.z + v.w;
    #pragma unroll
    for (int d = 1; d < 32; d <<= 1) s += __shfl_xor(s, d, 32);
    if ((t & 31) == 0) red[t >> 5] = s;
    __syncthreads();
    float tot = 0.f;
    #pragma unroll
    for (int w = 0; w < 8; w++) tot += red[w];
    const float mu = tot * (1.0f / DMODEL);

    float dx = v.x - mu, dy = v.y - mu, dz = v.z - mu, dw = v.w - mu;
    float sq = dx*dx + dy*dy + dz*dz + dw*dw;
    #pragma unroll
    for (int d = 1; d < 32; d <<= 1) sq += __shfl_xor(sq, d, 32);
    if ((t & 31) == 0) red[8 + (t >> 5)] = sq;
    __syncthreads();
    float tot2 = 0.f;
    #pragma unroll
    for (int w = 0; w < 8; w++) tot2 += red[8 + w];
    const float rstd = rsqrtf(tot2 * (1.0f / DMODEL) + 1e-5f);

    const int c0 = t * 4;
    bf16_t* o = out + (size_t)row * DMODEL + c0;
    o[0] = (bf16_t)(dx * rstd * g[c0 + 0] + be[c0 + 0]);
    o[1] = (bf16_t)(dy * rstd * g[c0 + 1] + be[c0 + 1]);
    o[2] = (bf16_t)(dz * rstd * g[c0 + 2] + be[c0 + 2]);
    o[3] = (bf16_t)(dw * rstd * g[c0 + 3] + be[c0 + 3]);
}

// ---------------------------------------------- weight transpose + bf16 cast --
// Wt[n*K + k] = bf16(W[k*N + n]);  32x32 LDS tile, coalesced both sides.
__global__ __launch_bounds__(256)
void transpose_cvt_kernel(const float* __restrict__ W, bf16_t* __restrict__ Wt,
                          int K, int N)
{
    __shared__ bf16_t tile[32][33];
    const int t  = threadIdx.x;
    const int tx = t & 31, ty = t >> 5;
    const int k0 = blockIdx.y * 32, n0 = blockIdx.x * 32;
    #pragma unroll
    for (int r = ty; r < 32; r += 8)
        tile[tx][r] = (bf16_t)W[(size_t)(k0 + r) * N + n0 + tx];
    __syncthreads();
    #pragma unroll
    for (int r = ty; r < 32; r += 8)
        Wt[(size_t)(n0 + r) * K + k0 + tx] = tile[r][tx];
}

// -------------------------------------------------------------------- GEMM ---
// C[M,N] = act(A_bf16[M,K] * Wt_bf16[N,K]^T + bias) (+resid).
// Block tile 128x128 (8 waves, wave tile 32x64), K-step 32, TDM double-buffer.
template<bool RELU, bool RESID, bool OUT_BF16>
__global__ __launch_bounds__(256)
void gemm_bf16_kernel(const bf16_t* __restrict__ A, const bf16_t* __restrict__ Wt,
                      const float* __restrict__ bias, const float* __restrict__ resid,
                      void* __restrict__ outp, int M, int N, int K)
{
    __shared__ __align__(16) bf16_t ldsA[2][128][40];   // 128x32 + TDM pad
    __shared__ __align__(16) bf16_t ldsB[2][128][40];   // 128x32 of Wt rows

    const int t    = threadIdx.x;
    const int wave = t >> 5;
    const int lane = t & 31;
    const int half = lane >> 4;
    const int lr   = lane & 15;
    const int m0   = blockIdx.y * 128;
    const int n0   = blockIdx.x * 128;
    const int mw   = (wave >> 1) * 32;   // wave rows within block tile
    const int nw   = (wave & 1) * 64;    // wave cols within block tile

    v8f acc[2][4];
    #pragma unroll
    for (int mt = 0; mt < 2; mt++)
        #pragma unroll
        for (int nt = 0; nt < 4; nt++)
            #pragma unroll
            for (int j = 0; j < 8; j++) acc[mt][nt][j] = 0.f;

    const int nsteps = K >> 5;

#if HAVE_TDM
    if (wave == 0) {   // pad: every 16 dwords (64B row) insert 4 dwords -> stride 40
        tdm_load_2d(A  + (size_t)m0 * K, &ldsA[0][0][0], 32, 128, (unsigned)K, 3, 3);
        tdm_load_2d(Wt + (size_t)n0 * K, &ldsB[0][0][0], 32, 128, (unsigned)K, 3, 3);
    }
#endif
    for (int s = 0; s < nsteps; ++s) {
        const int buf = s & 1;
#if HAVE_TDM
        if (s + 1 < nsteps) {
            if (wave == 0) {
                tdm_load_2d(A  + (size_t)m0 * K + (size_t)(s + 1) * 32,
                            &ldsA[buf ^ 1][0][0], 32, 128, (unsigned)K, 3, 3);
                tdm_load_2d(Wt + (size_t)n0 * K + (size_t)(s + 1) * 32,
                            &ldsB[buf ^ 1][0][0], 32, 128, (unsigned)K, 3, 3);
                __builtin_amdgcn_s_wait_tensorcnt(2);   // oldest pair (step s) done
            }
        } else {
            if (wave == 0) __builtin_amdgcn_s_wait_tensorcnt(0);
        }
#else
        stage_tile(&ldsA[buf][0][0], 40, 16, A  + (size_t)m0 * K + (size_t)s * 32, K, 128, t);
        stage_tile(&ldsB[buf][0][0], 40, 16, Wt + (size_t)n0 * K + (size_t)s * 32, K, 128, t);
#endif
        __syncthreads();

        Frag fa[2];
        #pragma unroll
        for (int mt = 0; mt < 2; mt++) {
            fa[mt].q[0] = *(const uint4*)&ldsA[buf][mw + mt * 16 + lr][8 * half];
            fa[mt].q[1] = *(const uint4*)&ldsA[buf][mw + mt * 16 + lr][16 + 8 * half];
        }
        #pragma unroll
        for (int nt = 0; nt < 4; nt++) {
            Frag fb;
            fb.q[0] = *(const uint4*)&ldsB[buf][nw + nt * 16 + lr][8 * half];
            fb.q[1] = *(const uint4*)&ldsB[buf][nw + nt * 16 + lr][16 + 8 * half];
            #pragma unroll
            for (int mt = 0; mt < 2; mt++)
                acc[mt][nt] = wmma_bf16(fa[mt], fb, acc[mt][nt]);
        }
        __syncthreads();
    }

    // Epilogue: lane holds col (nw+nt*16+lr), rows j + 8*half (+16*mt).
    #pragma unroll
    for (int nt = 0; nt < 4; nt++) {
        const int col = n0 + nw + nt * 16 + lr;
        const float bv = bias[col];
        #pragma unroll
        for (int mt = 0; mt < 2; mt++) {
            #pragma unroll
            for (int j = 0; j < 8; j++) {
                const int row = m0 + mw + mt * 16 + j + 8 * half;
                float v = acc[mt][nt][j] + bv;
                if (RELU)  v = v > 0.f ? v : 0.f;
                if (RESID) v += resid[(size_t)row * N + col];
                if (OUT_BF16) ((bf16_t*)outp)[(size_t)row * N + col] = (bf16_t)v;
                else          ((float*) outp)[(size_t)row * N + col] = v;
            }
        }
    }
}

// ------------------------------------------------------- Flash attention -----
// Block = 8 waves x 16 query rows = 128 q rows of one (b,h). Shared 64-key
// tiles: K staged by TDM ([key][dk]), V staged transposed ([dk][key]).
__global__ __launch_bounds__(256)
void attn_kernel(const bf16_t* __restrict__ Q, const bf16_t* __restrict__ Km,
                 const bf16_t* __restrict__ V, bf16_t* __restrict__ O)
{
    __shared__ __align__(16) bf16_t ldsK [64][72];      // [key][dk], TDM pad
    __shared__ __align__(16) bf16_t ldsVt[64][72];      // [dk][key]
    __shared__ __align__(16) bf16_t ldsP [8][16][72];   // per-wave P transpose

    const int t    = threadIdx.x;
    const int wave = t >> 5;
    const int lane = t & 31;
    const int half = lane >> 4;
    const int lr   = lane & 15;
    const int b    = blockIdx.z;
    const int h    = blockIdx.y;
    const int qbase = blockIdx.x * 128 + wave * 16;

    const size_t rowQ = (size_t)(b * SEQ + qbase + lr) * DMODEL + h * DKH;
    Frag qf[2];
    #pragma unroll
    for (int ks = 0; ks < 2; ks++) {
        qf[ks].q[0] = *(const uint4*)(Q + rowQ + 32 * ks + 8 * half);
        qf[ks].q[1] = *(const uint4*)(Q + rowQ + 32 * ks + 16 + 8 * half);
    }

    v8f ctx[4];
    float mrow[8], lrow[8];
    #pragma unroll
    for (int j = 0; j < 8; j++) { mrow[j] = -1e30f; lrow[j] = 0.f; }
    #pragma unroll
    for (int nt = 0; nt < 4; nt++)
        #pragma unroll
        for (int j = 0; j < 8; j++) ctx[nt][j] = 0.f;

    const float scale = 0.125f;                    // 1/sqrt(64)
    const int kend_wave = qbase + 15;
    const int kend_blk  = blockIdx.x * 128 + 127;  // block-shared causal bound

    for (int kb = 0; kb <= kend_blk; kb += 64) {
        const bf16_t* ksrc = Km + (size_t)(b * SEQ + kb) * DMODEL + h * DKH;
#if HAVE_TDM
        if (wave == 0)   // 128B rows (32 dw), pad 4 dw -> stride 72 elements
            tdm_load_2d(ksrc, &ldsK[0][0], 64, 64, DMODEL, 4, 3);
#else
        stage_tile(&ldsK[0][0], 72, 32, ksrc, DMODEL, 64, t);
#endif
        // V tile transposed: read coalesced over dk, scatter to ldsVt[d][key].
        for (int i = t; i < 64 * 64; i += 256) {
            const int key = i >> 6, d = i & 63;
            ldsVt[d][key] = V[(size_t)(b * SEQ + kb + key) * DMODEL + h * DKH + d];
        }
#if HAVE_TDM
        if (wave == 0) __builtin_amdgcn_s_wait_tensorcnt(0);
#endif
        __syncthreads();

        if (kb <= kend_wave) {   // wave-uniform: EXEC stays all-ones for WMMA
            // ---- scores S = Q K^T (16x64) ----
            v8f sc[4];
            #pragma unroll
            for (int nt = 0; nt < 4; nt++)
                #pragma unroll
                for (int j = 0; j < 8; j++) sc[nt][j] = 0.f;
            #pragma unroll
            for (int nt = 0; nt < 4; nt++) {
                #pragma unroll
                for (int ks = 0; ks < 2; ks++) {
                    Frag fk;   // lane = key index, K-dim = dk
                    fk.q[0] = *(const uint4*)&ldsK[nt * 16 + lr][32 * ks + 8 * half];
                    fk.q[1] = *(const uint4*)&ldsK[nt * 16 + lr][32 * ks + 16 + 8 * half];
                    sc[nt] = wmma_bf16(qf[ks], fk, sc[nt]);
                }
            }
            // ---- online softmax (rows j + 8*half across 16 lanes) ----
            #pragma unroll
            for (int j = 0; j < 8; j++) {
                const int q = qbase + j + 8 * half;
                float s[4];
                #pragma unroll
                for (int nt = 0; nt < 4; nt++) {
                    const int key = kb + nt * 16 + lr;
                    s[nt] = (key <= q) ? sc[nt][j] * scale : -1e30f;
                }
                float mx = fmaxf(fmaxf(s[0], s[1]), fmaxf(s[2], s[3]));
                #pragma unroll
                for (int d = 1; d < 16; d <<= 1) mx = fmaxf(mx, __shfl_xor(mx, d, 32));
                const float mnew = fmaxf(mrow[j], mx);
                float p[4], rs = 0.f;
                #pragma unroll
                for (int nt = 0; nt < 4; nt++) { p[nt] = __expf(s[nt] - mnew); rs += p[nt]; }
                #pragma unroll
                for (int d = 1; d < 16; d <<= 1) rs += __shfl_xor(rs, d, 32);
                const float corr = __expf(mrow[j] - mnew);
                lrow[j] = lrow[j] * corr + rs;
                mrow[j] = mnew;
                #pragma unroll
                for (int nt = 0; nt < 4; nt++) ctx[nt][j] *= corr;
                #pragma unroll
                for (int nt = 0; nt < 4; nt++)
                    ldsP[wave][j + 8 * half][nt * 16 + lr] = (bf16_t)p[nt];
            }
            // ---- ctx += P V ----
            #pragma unroll
            for (int ks = 0; ks < 2; ks++) {
                Frag fp;
                fp.q[0] = *(const uint4*)&ldsP[wave][lr][32 * ks + 8 * half];
                fp.q[1] = *(const uint4*)&ldsP[wave][lr][32 * ks + 16 + 8 * half];
                #pragma unroll
                for (int nt = 0; nt < 4; nt++) {
                    Frag fv;   // lane = dk col, K-dim = key
                    fv.q[0] = *(const uint4*)&ldsVt[nt * 16 + lr][32 * ks + 8 * half];
                    fv.q[1] = *(const uint4*)&ldsVt[nt * 16 + lr][32 * ks + 16 + 8 * half];
                    ctx[nt] = wmma_bf16(fp, fv, ctx[nt]);
                }
            }
        }
        __syncthreads();
    }

    #pragma unroll
    for (int nt = 0; nt < 4; nt++) {
        #pragma unroll
        for (int j = 0; j < 8; j++) {
            const int q = qbase + j + 8 * half;
            const float v = ctx[nt][j] / lrow[j];
            O[(size_t)(b * SEQ + q) * DMODEL + h * DKH + nt * 16 + lr] = (bf16_t)v;
        }
    }
}

// ------------------------------------------------------------- host driver ---
extern "C" void kernel_launch(void* const* d_in, const int* in_sizes, int n_in,
                              void* d_out, int out_size, void* d_ws, size_t ws_size,
                              hipStream_t stream)
{
    const float* x   = (const float*)d_in[0];
    const float* Wq  = (const float*)d_in[1];
    const float* bq  = (const float*)d_in[2];
    const float* Wk  = (const float*)d_in[3];
    const float* bk  = (const float*)d_in[4];
    const float* Wv  = (const float*)d_in[5];
    const float* bv  = (const float*)d_in[6];
    const float* Wo  = (const float*)d_in[7];
    const float* bo  = (const float*)d_in[8];
    const float* W1  = (const float*)d_in[9];
    const float* b1  = (const float*)d_in[10];
    const float* W2  = (const float*)d_in[11];
    const float* b2  = (const float*)d_in[12];
    const float* g1  = (const float*)d_in[13];
    const float* be1 = (const float*)d_in[14];
    const float* g2  = (const float*)d_in[15];
    const float* be2 = (const float*)d_in[16];
    float* out = (float*)d_out;
    (void)in_sizes; (void)n_in; (void)out_size; (void)ws_size;

    char* ws = (char*)d_ws;
    size_t off = 0;
    auto alloc = [&](size_t bytes) -> char* {
        char* p = ws + off;
        off = (off + bytes + 255) & ~(size_t)255;
        return p;
    };
    const size_t MD = (size_t)MROWS * DMODEL;
    bf16_t* h1   = (bf16_t*)alloc(MD * 2);
    bf16_t* Qb   = (bf16_t*)alloc(MD * 2);
    bf16_t* Kb   = (bf16_t*)alloc(MD * 2);
    bf16_t* Vb   = (bf16_t*)alloc(MD * 2);
    bf16_t* ctxb = (bf16_t*)alloc(MD * 2);
    float*  x1   = (float*) alloc(MD * 4);
    bf16_t* h2   = (bf16_t*)alloc(MD * 2);
    bf16_t* a1   = (bf16_t*)alloc((size_t)MROWS * DFF * 2);
    bf16_t* Wqt  = (bf16_t*)alloc((size_t)DMODEL * DMODEL * 2);  // [N][K]
    bf16_t* Wkt  = (bf16_t*)alloc((size_t)DMODEL * DMODEL * 2);
    bf16_t* Wvt  = (bf16_t*)alloc((size_t)DMODEL * DMODEL * 2);
    bf16_t* Wot  = (bf16_t*)alloc((size_t)DMODEL * DMODEL * 2);
    bf16_t* W1t  = (bf16_t*)alloc((size_t)DFF * DMODEL * 2);     // [4096][1024]
    bf16_t* W2t  = (bf16_t*)alloc((size_t)DMODEL * DFF * 2);     // [1024][4096]

    const dim3 blk(256);

    // 0) weights -> bf16, transposed to [N][K] (TDM-friendly contiguous rows)
    transpose_cvt_kernel<<<dim3(DMODEL/32, DMODEL/32), blk, 0, stream>>>(Wq, Wqt, DMODEL, DMODEL);
    transpose_cvt_kernel<<<dim3(DMODEL/32, DMODEL/32), blk, 0, stream>>>(Wk, Wkt, DMODEL, DMODEL);
    transpose_cvt_kernel<<<dim3(DMODEL/32, DMODEL/32), blk, 0, stream>>>(Wv, Wvt, DMODEL, DMODEL);
    transpose_cvt_kernel<<<dim3(DMODEL/32, DMODEL/32), blk, 0, stream>>>(Wo, Wot, DMODEL, DMODEL);
    transpose_cvt_kernel<<<dim3(DFF/32,    DMODEL/32), blk, 0, stream>>>(W1, W1t, DMODEL, DFF);
    transpose_cvt_kernel<<<dim3(DMODEL/32, DFF/32),    blk, 0, stream>>>(W2, W2t, DFF, DMODEL);

    const dim3 gD(DMODEL / 128, MROWS / 128);   // N=1024 GEMMs
    const dim3 gF(DFF / 128,    MROWS / 128);   // N=4096 GEMM

    // 1) h1 = LN1(x)
    ln_bf16_kernel<<<MROWS, blk, 0, stream>>>(x, g1, be1, h1);
    // 2) Q,K,V projections
    gemm_bf16_kernel<false, false, true><<<gD, blk, 0, stream>>>(h1, Wqt, bq, nullptr, Qb, MROWS, DMODEL, DMODEL);
    gemm_bf16_kernel<false, false, true><<<gD, blk, 0, stream>>>(h1, Wkt, bk, nullptr, Kb, MROWS, DMODEL, DMODEL);
    gemm_bf16_kernel<false, false, true><<<gD, blk, 0, stream>>>(h1, Wvt, bv, nullptr, Vb, MROWS, DMODEL, DMODEL);
    // 3) causal flash attention
    attn_kernel<<<dim3(SEQ / 128, NHEADS, BATCH), blk, 0, stream>>>(Qb, Kb, Vb, ctxb);
    // 4) x1 = x + ctx @ Wo + bo
    gemm_bf16_kernel<false, true, false><<<gD, blk, 0, stream>>>(ctxb, Wot, bo, x, x1, MROWS, DMODEL, DMODEL);
    // 5) h2 = LN2(x1)
    ln_bf16_kernel<<<MROWS, blk, 0, stream>>>(x1, g2, be2, h2);
    // 6) a1 = relu(h2 @ W1 + b1)
    gemm_bf16_kernel<true, false, true><<<gF, blk, 0, stream>>>(h2, W1t, b1, nullptr, a1, MROWS, DFF, DMODEL);
    // 7) out = x1 + a1 @ W2 + b2
    gemm_bf16_kernel<false, true, false><<<gD, blk, 0, stream>>>(a1, W2t, b2, x1, out, MROWS, DMODEL, DFF);
}
